// Decoder_4587025072452
// MI455X (gfx1250) — matrix operand order repaired and checked by reading, not compiled, not run on hardware
//
#include <hip/hip_runtime.h>
#include <hip/hip_bf16.h>
#include <math.h>

// ---------------------------------------------------------------------------
// Problem constants (from reference): V=32000 E=512 H=1024 B=128 S=256 L=2
// ---------------------------------------------------------------------------
#define VV 32000
#define EE 512
#define HH 1024
#define BB 128
#define SS 256

typedef __bf16 bf16_t;
typedef bf16_t v16bf __attribute__((ext_vector_type(16)));
typedef float  v8f   __attribute__((ext_vector_type(8)));

// ---------------------------------------------------------------------------
// WMMA inner loop: accumulate A[M=128,K] (fp32, row-major, lda) times
//   BT=true : W[N,K] row-major  (C = A * W^T)   -- weight rows contiguous
//   BT=false: W[K,N] row-major  (C = A * W)
// into a 16x16 f32 accumulator tile using v_wmma_f32_16x16x32_bf16,
// converting fp32 -> bf16 on the fly.
//
// Operand lane layouts used here (wave32, contiguous k, not the ISA table order):
//   A (16x32 bf16): lane L<16  -> row M=L,  K = ka..ka+15, ka=0
//                   lane L>=16 -> row M=L-16, same with ka=16
//   B (32x16 bf16): lane L<16  -> col N=L,  K = kb..kb+15, kb=0
//                   lane L>=16 -> col N=L-16, kb=16
// ---------------------------------------------------------------------------
template <bool BT>
__device__ __forceinline__ v8f wmma_gemm_acc(const float* __restrict__ A, int lda,
                                             const float* __restrict__ W, int ldw,
                                             int K, int m, int n, int ka, int kb,
                                             v8f acc)
{
    for (int k0 = 0; k0 < K; k0 += 32) {
        const float* ap = A + (size_t)m * lda + k0;
        v16bf a, b;
#pragma unroll
        for (int j = 0; j < 8; ++j) {
            a[j]     = (bf16_t)ap[ka + j];
            a[8 + j] = (bf16_t)ap[ka + 8 + j];
        }
        if (BT) {
            const float* bp = W + (size_t)n * ldw + (k0 + kb);
#pragma unroll
            for (int j = 0; j < 16; ++j) b[j] = (bf16_t)bp[j];
        } else {
            const float* bp = W + (size_t)(k0 + kb) * ldw + n;
#pragma unroll
            for (int j = 0; j < 16; ++j) b[j] = (bf16_t)bp[(size_t)j * ldw];
        }
        acc = __builtin_amdgcn_wmma_f32_16x16x32_bf16(
            /*neg_a=*/false, a, /*neg_b=*/false, b,
            /*c_mod=*/(short)0, acc, /*reuse_a=*/false, /*reuse_b=*/false);
    }
    return acc;
}

// ---------------------------------------------------------------------------
// Fused dual-input GEMM: C[M=128, N] = act( A1*W1^T + A2*W2^T + bias1 + bias2 )
// One 256-thread workgroup (8 wave32) covers the full M=128 for one 16-wide
// N strip (blockIdx.x), so weight rows stream from HBM once.
// act: 0 = none, 1 = tanh
// ---------------------------------------------------------------------------
template <bool BT>
__global__ __launch_bounds__(256)
void gemm_bt_kernel(const float* __restrict__ A1, int lda1,
                    const float* __restrict__ W1, int ldw1, int K1,
                    const float* __restrict__ A2, int lda2,
                    const float* __restrict__ W2, int ldw2, int K2,
                    const float* __restrict__ bias1,
                    const float* __restrict__ bias2,
                    float* __restrict__ C, int ldc, int act)
{
    const int lane = threadIdx.x & 31;
    const int wave = threadIdx.x >> 5;           // 0..7 -> M tile
    const int n0   = blockIdx.x * 16;
    const int m    = wave * 16 + (lane & 15);
    const int n    = n0 + (lane & 15);
    const int ka   = (lane < 16) ? 0 : 16;       // A half-K select
    const int kb   = (lane < 16) ? 0 : 16;       // B K-half select

    v8f acc = {0.f, 0.f, 0.f, 0.f, 0.f, 0.f, 0.f, 0.f};
    acc = wmma_gemm_acc<BT>(A1, lda1, W1, ldw1, K1, m, n, ka, kb, acc);
    if (A2) acc = wmma_gemm_acc<BT>(A2, lda2, W2, ldw2, K2, m, n, ka, kb, acc);

    const int ccol = n0 + (lane & 15);
    float bs = 0.f;
    if (bias1) bs += bias1[ccol];
    if (bias2) bs += bias2[ccol];

    // D layout: VGPR r -> row (lane<16 ? r : r+8) of the 16-row tile
    const int crow = wave * 16 + ((lane < 16) ? 0 : 8);
#pragma unroll
    for (int r = 0; r < 8; ++r) {
        float v = acc[r] + bs;
        if (act == 1) v = tanhf(v);
        C[(size_t)(crow + r) * ldc + ccol] = v;
    }
}

// ---------------------------------------------------------------------------
// Embedding gather: x[b,e] = emb[input_ids[b], e]
// ---------------------------------------------------------------------------
__global__ __launch_bounds__(256)
void gather_emb_kernel(const int* __restrict__ ids,
                       const float* __restrict__ emb,
                       float* __restrict__ x)
{
    int idx = blockIdx.x * 256 + threadIdx.x;    // B*E = 65536
    if (idx >= BB * EE) return;
    int b = idx >> 9;          // /E
    int e = idx & (EE - 1);
    x[idx] = emb[(size_t)ids[b] * EE + e];
}

// ---------------------------------------------------------------------------
// LSTM pointwise: gates[B,4H] (PyTorch order i,f,g,o) + c_prev -> h,c
// ---------------------------------------------------------------------------
__global__ __launch_bounds__(256)
void lstm_elem_kernel(const float* __restrict__ gates,
                      const float* __restrict__ c_prev,
                      float* __restrict__ h_out,
                      float* __restrict__ c_out)
{
    int idx = blockIdx.x * 256 + threadIdx.x;    // B*H = 131072
    if (idx >= BB * HH) return;
    int b = idx >> 10;
    int j = idx & (HH - 1);
    const float* g = gates + (size_t)b * (4 * HH);
    float gi = g[j];
    float gf = g[HH + j];
    float gg = g[2 * HH + j];
    float go = g[3 * HH + j];
    float si = 1.f / (1.f + expf(-gi));
    float sf = 1.f / (1.f + expf(-gf));
    float so = 1.f / (1.f + expf(-go));
    float c  = sf * c_prev[idx] + si * tanhf(gg);
    float h  = so * tanhf(c);
    c_out[idx] = c;
    h_out[idx] = h;
}

// ---------------------------------------------------------------------------
// Attention (one workgroup of 256 thr per batch element b):
//   t       = attn_b . rnn[b]
//   score[s]= enc[s,b,:] . q[b,:] + t
//   attn    = softmax(score)
//   ctx[b,:]= sum_s attn[s] * enc[s,b,:]
// enc slice per b is 1 MB -> pass 2 hits L2 (enc total 134 MB < 192 MB L2).
// ---------------------------------------------------------------------------
__global__ __launch_bounds__(256)
void attention_kernel(const float* __restrict__ enc,   // [S,B,H]
                      const float* __restrict__ q,     // [B,H]
                      const float* __restrict__ rnn,   // [B,H]
                      const float* __restrict__ attn_b,// [H]
                      float* __restrict__ context)     // [B,H]
{
    __shared__ float sq[HH];
    __shared__ float sc[SS];
    __shared__ float redA[8];
    __shared__ float redB[8];

    const int b    = blockIdx.x;
    const int tid  = threadIdx.x;
    const int lane = tid & 31;
    const int wave = tid >> 5;

    // stage q, and accumulate t = attn_b . rnn[b]
    float tpart = 0.f;
    for (int h = tid; h < HH; h += 256) {
        float qv = q[(size_t)b * HH + h];
        sq[h] = qv;
        tpart += attn_b[h] * rnn[(size_t)b * HH + h];
    }
#pragma unroll
    for (int off = 16; off; off >>= 1) tpart += __shfl_down(tpart, off, 32);
    if (lane == 0) redA[wave] = tpart;
    __syncthreads();
    float tb = 0.f;
#pragma unroll
    for (int w = 0; w < 8; ++w) tb += redA[w];

    // scores: each wave owns strided s values
    for (int s = wave; s < SS; s += 8) {
        const float* ep = enc + ((size_t)s * BB + b) * HH;
        float p = 0.f;
        for (int h = lane; h < HH; h += 32) p += ep[h] * sq[h];
#pragma unroll
        for (int off = 16; off; off >>= 1) p += __shfl_down(p, off, 32);
        if (lane == 0) sc[s] = p + tb;
    }
    __syncthreads();

    // softmax over 256 scores; thread tid owns s = tid
    float sv = sc[tid];
    float mx = sv;
#pragma unroll
    for (int off = 16; off; off >>= 1) mx = fmaxf(mx, __shfl_down(mx, off, 32));
    if (lane == 0) redA[wave] = mx;
    __syncthreads();
    float bm = redA[0];
#pragma unroll
    for (int w = 1; w < 8; ++w) bm = fmaxf(bm, redA[w]);

    float ev = expf(sv - bm);
    float sp = ev;
#pragma unroll
    for (int off = 16; off; off >>= 1) sp += __shfl_down(sp, off, 32);
    if (lane == 0) redB[wave] = sp;
    __syncthreads();
    float tot = 0.f;
#pragma unroll
    for (int w = 0; w < 8; ++w) tot += redB[w];

    sc[tid] = ev / tot;
    __syncthreads();

    // context: coalesced second pass over enc slice
    for (int h = tid; h < HH; h += 256) {
        float acc = 0.f;
        for (int s = 0; s < SS; ++s)
            acc += sc[s] * enc[((size_t)s * BB + b) * HH + h];
        context[(size_t)b * HH + h] = acc;
    }
}

// ---------------------------------------------------------------------------
// Launch: graph-capture-safe, everything on `stream`.
// d_out layout (fp32): logits[128*32000] | h_new[2*128*1024] | c_new[2*128*1024]
// ---------------------------------------------------------------------------
extern "C" void kernel_launch(void* const* d_in, const int* in_sizes, int n_in,
                              void* d_out, int out_size, void* d_ws, size_t ws_size,
                              hipStream_t stream)
{
    (void)in_sizes; (void)n_in; (void)out_size; (void)ws_size;

    const int*   input_ids = (const int*)  d_in[0];
    const float* h0        = (const float*)d_in[1];   // [2,B,H]
    const float* c0        = (const float*)d_in[2];   // [2,B,H]
    const float* enc       = (const float*)d_in[3];   // [S,B,H]
    const float* emb       = (const float*)d_in[4];   // [V,E]
    const float* w_ih0     = (const float*)d_in[5];   // [4H,E]
    const float* w_hh0     = (const float*)d_in[6];   // [4H,H]
    const float* b_ih0     = (const float*)d_in[7];
    const float* b_hh0     = (const float*)d_in[8];
    const float* w_ih1     = (const float*)d_in[9];   // [4H,H]
    const float* w_hh1     = (const float*)d_in[10];  // [4H,H]
    const float* b_ih1     = (const float*)d_in[11];
    const float* b_hh1     = (const float*)d_in[12];
    const float* attn_w    = (const float*)d_in[13];  // [H,H]
    const float* attn_b    = (const float*)d_in[14];  // [H]
    const float* concat_w  = (const float*)d_in[15];  // [H,2H]
    const float* concat_b  = (const float*)d_in[16];  // [H]
    const float* out_w     = (const float*)d_in[17];  // [V,H]
    const float* out_b     = (const float*)d_in[18];  // [V]

    float* out    = (float*)d_out;
    float* logits = out;                               // [B,V]
    float* h_new  = out + (size_t)BB * VV;             // [2,B,H]
    float* c_new  = h_new + 2 * (size_t)BB * HH;       // [2,B,H]
    float* h_0 = h_new;
    float* h_1 = h_new + (size_t)BB * HH;
    float* c_0 = c_new;
    float* c_1 = c_new + (size_t)BB * HH;

    float* ws      = (float*)d_ws;
    float* x       = ws;                               // [B,E]      65536
    float* gates   = x + (size_t)BB * EE;              // [B,4H]    524288 (reused)
    float* qbuf    = gates + (size_t)BB * 4 * HH;      // [B,H]     131072
    float* ctx     = qbuf + (size_t)BB * HH;           // [B,H]     131072
    float* cat_out = ctx + (size_t)BB * HH;            // [B,H]     131072

    // 1) x = emb[input_ids]
    gather_emb_kernel<<<(BB * EE) / 256, 256, 0, stream>>>(input_ids, emb, x);

    // 2) gates0 = x @ w_ih0^T + h0[0] @ w_hh0^T + b_ih0 + b_hh0
    gemm_bt_kernel<true><<<(4 * HH) / 16, 256, 0, stream>>>(
        x, EE, w_ih0, EE, EE,
        h0, HH, w_hh0, HH, HH,
        b_ih0, b_hh0, gates, 4 * HH, /*act=*/0);

    // 3) (h_0, c_0) = lstm(gates0, c0[0])
    lstm_elem_kernel<<<(BB * HH) / 256, 256, 0, stream>>>(gates, c0, h_0, c_0);

    // 4) gates1 = h_0 @ w_ih1^T + h0[1] @ w_hh1^T + b_ih1 + b_hh1
    gemm_bt_kernel<true><<<(4 * HH) / 16, 256, 0, stream>>>(
        h_0, HH, w_ih1, HH, HH,
        h0 + (size_t)BB * HH, HH, w_hh1, HH, HH,
        b_ih1, b_hh1, gates, 4 * HH, /*act=*/0);

    // 5) (h_1, c_1) = lstm(gates1, c0[1])
    lstm_elem_kernel<<<(BB * HH) / 256, 256, 0, stream>>>(
        gates, c0 + (size_t)BB * HH, h_1, c_1);

    // 6) q = h_1 @ attn_w   (attn_w used non-transposed: q[b,k]=sum_h rnn[b,h]*attn_w[h,k])
    gemm_bt_kernel<false><<<HH / 16, 256, 0, stream>>>(
        h_1, HH, attn_w, HH, HH,
        nullptr, 0, nullptr, 0, 0,
        nullptr, nullptr, qbuf, HH, /*act=*/0);

    // 7) scores -> softmax -> context (factored Luong attention)
    attention_kernel<<<BB, 256, 0, stream>>>(enc, qbuf, h_1, attn_b, ctx);

    // 8) concat_out = tanh([h_1 | ctx] @ concat_w^T + concat_b)  (dual-A split of K=2H)
    gemm_bt_kernel<true><<<HH / 16, 256, 0, stream>>>(
        h_1, HH, concat_w, 2 * HH, HH,
        ctx, HH, concat_w + HH, 2 * HH, HH,
        concat_b, nullptr, cat_out, HH, /*act=*/1);

    // 9) logits = concat_out @ out_w^T + out_b
    gemm_bt_kernel<true><<<VV / 16, 256, 0, stream>>>(
        cat_out, HH, out_w, HH, HH,
        nullptr, 0, nullptr, 0, 0,
        out_b, nullptr, logits, VV, /*act=*/0);
}
